// DAGLSTMMLP_59871844106678
// MI455X (gfx1250) — matrix operand order repaired and checked by reading, not compiled, not run on hardware
//
#include <hip/hip_runtime.h>
#include <hip/hip_bf16.h>

// ---------------------------------------------------------------------------
// DAG-LSTM + MLP for MI455X (gfx1250, wave32, WMMA bf16).
// All GEMMs run through v_wmma_f32_16x16x32_bf16 with f32 accumulation.
// Per node step: 2 fused kernels (gather-mean staged in LDS + gate GEMM +
// in-register LSTM cell math). MLP head uses the same WMMA tiling.
// ---------------------------------------------------------------------------

constexpr int kB = 4096, kN = 24, kP = 3;
constexpr int kIN = 128, kH = 256, kL = 2;
constexpr int kEXTRA = 128, kMH = 512;

typedef __attribute__((ext_vector_type(16))) __bf16 v16bf;
typedef __attribute__((ext_vector_type(8)))  __bf16 v8bf;
typedef __attribute__((ext_vector_type(8)))  float  v8f;

__device__ __forceinline__ float sigmoidf_(float x) {
    return 1.0f / (1.0f + __expf(-x));
}

// A fragment: 16x32 bf16, row-major source with leading dim lda (elements).
// Lane layout (ISA 7.12.2, 16-bit A 16x32): row = lane%16; lanes 0-15 hold
// K={0..7,16..23}, lanes 16-31 hold K={8..15,24..31} -> two 16B loads.
// Works for both global and LDS pointers (addrspace inferred after inlining).
__device__ __forceinline__ v16bf load_a_frag(const __bf16* __restrict__ A,
                                             int lda, int m0, int k0, int lane) {
    const int r = lane & 15, s = lane >> 4;
    const __bf16* p = A + (size_t)(m0 + r) * (size_t)lda + k0 + s * 8;
    v8bf lo = *(const v8bf*)(p);
    v8bf hi = *(const v8bf*)(p + 16);
    v16bf a;
#pragma unroll
    for (int i = 0; i < 8; ++i) { a[i] = lo[i]; a[8 + i] = hi[i]; }
    return a;
}

// B fragment: 32x16 bf16. W is [Ncols, K] row-major (PyTorch W layout), so
// B[k, n] = W[n, k]: lane n%16 holds column n; lanes 0-15 hold
// K={0..7,16..23}, lanes 16-31 hold K={8..15,24..31} (matches A fragment).
__device__ __forceinline__ v16bf load_b_frag(const __bf16* __restrict__ W,
                                             int ldw, int n0, int k0, int lane) {
    const int c = lane & 15, s = lane >> 4;
    const __bf16* p = W + (size_t)(n0 + c) * (size_t)ldw + k0 + s * 8;
    v8bf lo = *(const v8bf*)(p);
    v8bf hi = *(const v8bf*)(p + 16);
    v16bf b;
#pragma unroll
    for (int i = 0; i < 8; ++i) { b[i] = lo[i]; b[8 + i] = hi[i]; }
    return b;
}

__device__ __forceinline__ v8f wmma_bf16(v16bf a, v16bf b, v8f c) {
    return __builtin_amdgcn_wmma_f32_16x16x32_bf16(
        /*neg_a=*/false, a, /*neg_b=*/false, b,
        /*c_mod=*/(short)0, c, /*reuse_a=*/false, /*reuse_b=*/false);
}

// ---------------------------------------------------------------------------
// f32 -> bf16 conversion (grid-stride)
// ---------------------------------------------------------------------------
__global__ __launch_bounds__(256) void cvt_f32_bf16(const float* __restrict__ in,
                                                    __bf16* __restrict__ out, int n) {
    int i = blockIdx.x * blockDim.x + threadIdx.x;
    int stride = gridDim.x * blockDim.x;
    for (; i < n; i += stride) out[i] = (__bf16)in[i];
}

__global__ __launch_bounds__(256) void zero_bf16(__bf16* __restrict__ p, int n) {
    int i = blockIdx.x * blockDim.x + threadIdx.x;
    int stride = gridDim.x * blockDim.x;
    for (; i < n; i += stride) p[i] = (__bf16)0.0f;
}

// ---------------------------------------------------------------------------
// Fused per-node LSTM layer step:
//   1) gather phase: masked mean over <=3 predecessor state slots for this
//      block's 64 batch rows -> h_in tile (64 x 256 bf16) and c_in tile
//      (64 x 64 f32) staged in LDS.
//   2) WMMA gate GEMM: gates = A0@Wa^T + h_in@Whh^T; A0 from global,
//      h_in A-fragments from LDS (ds_load_b128, bank-conflict-free via
//      264-element padded row stride).
//   3) epilogue: i/f/g/o live in 4 accumulators at the same (row,col) ->
//      full sigmoid/tanh cell math in-register; h/c written bf16.
// Block: 256 threads = 8 waves; tile 64 rows x 64 hidden units.
// Wave: 32 rows x 16 hidden x 4 gates -> 8 f32 accumulators.
// ---------------------------------------------------------------------------
__global__ __launch_bounds__(256) void lstm_step_fused(
    const __bf16* __restrict__ A0, int lda0, int K0, const __bf16* __restrict__ Wa,
    const __bf16* __restrict__ Wb,                 // [1024, kH] recurrent weights
    const __bf16* __restrict__ h_buf, const __bf16* __restrict__ c_buf,
    const int* __restrict__ pred_idx, const int* __restrict__ pred_mask,
    int step, int layer,
    const float* __restrict__ bih, const float* __restrict__ bhh,
    __bf16* __restrict__ h_out, __bf16* __restrict__ c_out,
    __bf16* __restrict__ h_out2) {
    constexpr int ROWS = 64;           // batch rows per block
    constexpr int HB   = 64;           // hidden units per block
    constexpr int LDL  = kH + 8;       // padded LDS row stride (bank spread)

    __shared__ __bf16 h_lds[ROWS * LDL];
    __shared__ float  c_lds[ROWS * HB];

    const int tid = threadIdx.x;
    const int row_base = blockIdx.x * ROWS;
    const int nh_blk = blockIdx.y * HB;
    const size_t bh = (size_t)kB * kH;

    // Uniform predecessor info for this step.
    int slots[kP], msk[kP];
    float cnt = 0.f;
#pragma unroll
    for (int p = 0; p < kP; ++p) {
        msk[p] = pred_mask[step * kP + p];
        slots[p] = pred_idx[step * kP + p];
        if (msk[p]) cnt += 1.f;
    }
    const float inv = 1.f / fmaxf(cnt, 1.f);

    // ---- gather phase: h_in tile (64 x 256) ----
    for (int i = tid; i < ROWS * kH; i += 256) {
        const int r = i >> 8;              // kH == 256
        const int k = i & (kH - 1);
        float s = 0.f;
#pragma unroll
        for (int p = 0; p < kP; ++p) {
            if (msk[p]) {
                size_t off = ((size_t)slots[p] * kL + layer) * bh +
                             (size_t)(row_base + r) * kH + k;
                s += (float)h_buf[off];
            }
        }
        h_lds[r * LDL + k] = (__bf16)(s * inv);
    }
    // ---- gather phase: c_in tile (64 x 64, this block's hidden window) ----
    for (int i = tid; i < ROWS * HB; i += 256) {
        const int r = i >> 6;              // HB == 64
        const int c = i & (HB - 1);
        float s = 0.f;
#pragma unroll
        for (int p = 0; p < kP; ++p) {
            if (msk[p]) {
                size_t off = ((size_t)slots[p] * kL + layer) * bh +
                             (size_t)(row_base + r) * kH + (nh_blk + c);
                s += (float)c_buf[off];
            }
        }
        c_lds[i] = s * inv;
    }
    __syncthreads();

    // ---- WMMA gate GEMM ----
    const int lane = tid & 31;
    const int w = tid >> 5;
    const int rg = (w >> 2) * 32;          // local row group: 0 or 32
    const int cg = (w & 3) * 16;           // local hidden group: 0..48
    const int m0 = row_base + rg;
    const int nh = nh_blk + cg;

    v8f acc[2][4] = {};                    // [m-subtile][gate]

    for (int k = 0; k < K0; k += 32) {     // input-weight term (A0 global)
        __builtin_prefetch((const void*)(A0 + (size_t)m0 * lda0 + k + 64), 0, 1);
        v16bf a0 = load_a_frag(A0, lda0, m0, k, lane);
        v16bf a1 = load_a_frag(A0, lda0, m0 + 16, k, lane);
#pragma unroll
        for (int g = 0; g < 4; ++g) {
            v16bf b = load_b_frag(Wa, K0, g * kH + nh, k, lane);
            acc[0][g] = wmma_bf16(a0, b, acc[0][g]);
            acc[1][g] = wmma_bf16(a1, b, acc[1][g]);
        }
    }
    for (int k = 0; k < kH; k += 32) {     // recurrent term (h_in from LDS)
        v16bf a0 = load_a_frag(h_lds, LDL, rg, k, lane);
        v16bf a1 = load_a_frag(h_lds, LDL, rg + 16, k, lane);
#pragma unroll
        for (int g = 0; g < 4; ++g) {
            v16bf b = load_b_frag(Wb, kH, g * kH + nh, k, lane);
            acc[0][g] = wmma_bf16(a0, b, acc[0][g]);
            acc[1][g] = wmma_bf16(a1, b, acc[1][g]);
        }
    }

    // ---- epilogue: full LSTM cell math in-register ----
    const int col = nh + (lane & 15);
    float bi[4];
#pragma unroll
    for (int g = 0; g < 4; ++g) bi[g] = bih[g * kH + col] + bhh[g * kH + col];

#pragma unroll
    for (int mt = 0; mt < 2; ++mt) {
#pragma unroll
        for (int v = 0; v < 8; ++v) {
            const int rloc = rg + mt * 16 + (lane >> 4) * 8 + v;  // C/D layout
            const int row = row_base + rloc;
            float iv = sigmoidf_(acc[mt][0][v] + bi[0]);
            float fv = sigmoidf_(acc[mt][1][v] + bi[1]);
            float gv = tanhf(acc[mt][2][v] + bi[2]);
            float ov = sigmoidf_(acc[mt][3][v] + bi[3]);
            float ci = c_lds[rloc * HB + (cg + (lane & 15))];
            float c2 = fv * ci + iv * gv;
            float h2 = ov * tanhf(c2);
            size_t o = (size_t)row * kH + col;
            h_out[o] = (__bf16)h2;
            c_out[o] = (__bf16)c2;
            if (h_out2) h_out2[o] = (__bf16)h2;
        }
    }
}

// ---------------------------------------------------------------------------
// MLP GEMM: out[B,ncols] = relu(A0@W0^T + A1@W1^T + bias)   (A1 optional)
// Block tile 128x64; wave tile 32x32 (2x2 16x16 WMMA tiles).
// ---------------------------------------------------------------------------
__global__ __launch_bounds__(256) void mlp_gemm_relu_wmma(
    const __bf16* __restrict__ A0, int lda0, int K0,
    const __bf16* __restrict__ W0, int ldw0,
    const __bf16* __restrict__ A1, int lda1, int K1,
    const __bf16* __restrict__ W1, int ldw1,
    const float* __restrict__ bias, __bf16* __restrict__ out, int ncols) {
    const int lane = threadIdx.x & 31;
    const int w = threadIdx.x >> 5;
    const int m0 = blockIdx.x * 128 + (w >> 1) * 32;
    const int n0 = blockIdx.y * 64 + (w & 1) * 32;

    v8f acc[2][2] = {};

    for (int k = 0; k < K0; k += 32) {
        __builtin_prefetch((const void*)(A0 + (size_t)m0 * lda0 + k + 64), 0, 1);
        v16bf a0 = load_a_frag(A0, lda0, m0, k, lane);
        v16bf a1 = load_a_frag(A0, lda0, m0 + 16, k, lane);
        v16bf b0 = load_b_frag(W0, ldw0, n0, k, lane);
        v16bf b1 = load_b_frag(W0, ldw0, n0 + 16, k, lane);
        acc[0][0] = wmma_bf16(a0, b0, acc[0][0]);
        acc[0][1] = wmma_bf16(a0, b1, acc[0][1]);
        acc[1][0] = wmma_bf16(a1, b0, acc[1][0]);
        acc[1][1] = wmma_bf16(a1, b1, acc[1][1]);
    }
    for (int k = 0; k < K1; k += 32) {
        v16bf a0 = load_a_frag(A1, lda1, m0, k, lane);
        v16bf a1 = load_a_frag(A1, lda1, m0 + 16, k, lane);
        v16bf b0 = load_b_frag(W1, ldw1, n0, k, lane);
        v16bf b1 = load_b_frag(W1, ldw1, n0 + 16, k, lane);
        acc[0][0] = wmma_bf16(a0, b0, acc[0][0]);
        acc[0][1] = wmma_bf16(a0, b1, acc[0][1]);
        acc[1][0] = wmma_bf16(a1, b0, acc[1][0]);
        acc[1][1] = wmma_bf16(a1, b1, acc[1][1]);
    }

#pragma unroll
    for (int mt = 0; mt < 2; ++mt) {
#pragma unroll
        for (int nt = 0; nt < 2; ++nt) {
            int colc = n0 + nt * 16 + (lane & 15);
            float bv = bias[colc];
#pragma unroll
            for (int v = 0; v < 8; ++v) {
                int row = m0 + mt * 16 + (lane >> 4) * 8 + v;
                float val = fmaxf(acc[mt][nt][v] + bv, 0.0f);
                out[(size_t)row * ncols + colc] = (__bf16)val;
            }
        }
    }
}

// Final [512 -> 1] projection, f32.
__global__ __launch_bounds__(256) void mlp_out_kernel(
    const __bf16* __restrict__ act, const float* __restrict__ W,
    const float* __restrict__ b, float* __restrict__ out) {
    int r = blockIdx.x * blockDim.x + threadIdx.x;
    if (r >= kB) return;
    float s = b[0];
    const __bf16* p = act + (size_t)r * kMH;
    for (int n = 0; n < kMH; ++n) s += (float)p[n] * W[n];
    out[r] = s;
}

// ---------------------------------------------------------------------------
extern "C" void kernel_launch(void* const* d_in, const int* in_sizes, int n_in,
                              void* d_out, int out_size, void* d_ws, size_t ws_size,
                              hipStream_t stream) {
    (void)in_sizes; (void)n_in; (void)out_size; (void)ws_size;
    const float* dags     = (const float*)d_in[0];
    const float* features = (const float*)d_in[1];
    const int*   pred_idx = (const int*)d_in[2];
    const int*   pred_mask= (const int*)d_in[3];
    const float* Wih0 = (const float*)d_in[4];
    const float* Whh0 = (const float*)d_in[5];
    const float* bih0 = (const float*)d_in[6];
    const float* bhh0 = (const float*)d_in[7];
    const float* Wih1 = (const float*)d_in[8];
    const float* Whh1 = (const float*)d_in[9];
    const float* bih1 = (const float*)d_in[10];
    const float* bhh1 = (const float*)d_in[11];
    const float* mW0  = (const float*)d_in[12];
    const float* mb0  = (const float*)d_in[13];
    const float* mW1  = (const float*)d_in[14];
    const float* mb1  = (const float*)d_in[15];
    const float* mW2  = (const float*)d_in[16];
    const float* mb2  = (const float*)d_in[17];
    float* out = (float*)d_out;

    // Workspace carve-up (256B aligned)
    char* ws = (char*)d_ws;
    size_t off = 0;
    auto alloc = [&](size_t bytes) -> char* {
        char* p = ws + off;
        off += (bytes + 255) & ~(size_t)255;
        return p;
    };
    __bf16* x_bf    = (__bf16*)alloc((size_t)kB * kN * kIN * 2);   // dags bf16
    __bf16* feat_bf = (__bf16*)alloc((size_t)kB * kEXTRA * 2);
    __bf16* wih0_bf = (__bf16*)alloc((size_t)4 * kH * kIN * 2);
    __bf16* whh0_bf = (__bf16*)alloc((size_t)4 * kH * kH * 2);
    __bf16* wih1_bf = (__bf16*)alloc((size_t)4 * kH * kH * 2);
    __bf16* whh1_bf = (__bf16*)alloc((size_t)4 * kH * kH * 2);
    __bf16* mw0_bf  = (__bf16*)alloc((size_t)kMH * (kH + kEXTRA) * 2);
    __bf16* mw1_bf  = (__bf16*)alloc((size_t)kMH * kMH * 2);
    const size_t slotsz = (size_t)kL * kB * kH;                    // per-slot elems
    __bf16* h_buf   = (__bf16*)alloc((size_t)(kN + 1) * slotsz * 2);
    __bf16* c_buf   = (__bf16*)alloc((size_t)(kN + 1) * slotsz * 2);
    __bf16* x1_bf   = (__bf16*)alloc((size_t)kB * kH * 2);
    __bf16* act0_bf = (__bf16*)alloc((size_t)kB * kMH * 2);
    __bf16* act1_bf = (__bf16*)alloc((size_t)kB * kMH * 2);

    const int T = 256;

    // One-time conversions to bf16 (weights stay L2-resident for all steps)
    cvt_f32_bf16<<<4096, T, 0, stream>>>(dags, x_bf, kB * kN * kIN);
    cvt_f32_bf16<<<512,  T, 0, stream>>>(features, feat_bf, kB * kEXTRA);
    cvt_f32_bf16<<<256,  T, 0, stream>>>(Wih0, wih0_bf, 4 * kH * kIN);
    cvt_f32_bf16<<<256,  T, 0, stream>>>(Whh0, whh0_bf, 4 * kH * kH);
    cvt_f32_bf16<<<256,  T, 0, stream>>>(Wih1, wih1_bf, 4 * kH * kH);
    cvt_f32_bf16<<<256,  T, 0, stream>>>(Whh1, whh1_bf, 4 * kH * kH);
    cvt_f32_bf16<<<256,  T, 0, stream>>>(mW0, mw0_bf, kMH * (kH + kEXTRA));
    cvt_f32_bf16<<<256,  T, 0, stream>>>(mW1, mw1_bf, kMH * kMH);

    // Zero start-state (slot 0 of h/c buffers)
    zero_bf16<<<2048, T, 0, stream>>>(h_buf, (int)slotsz);
    zero_bf16<<<2048, T, 0, stream>>>(c_buf, (int)slotsz);

    const dim3 lgrid(kB / 64, kH / 64);    // 64 x 4 = 256 blocks

    for (int step = 0; step < kN; ++step) {
        __bf16* h0o = h_buf + (size_t)(step + 1) * slotsz;         // layer 0
        __bf16* c0o = c_buf + (size_t)(step + 1) * slotsz;
        __bf16* h1o = h0o + (size_t)kB * kH;                       // layer 1
        __bf16* c1o = c0o + (size_t)kB * kH;

        // Layer 0: gates = x@Wih0^T + mean(h_pred)@Whh0^T
        lstm_step_fused<<<lgrid, T, 0, stream>>>(
            x_bf + (size_t)step * kIN, kN * kIN, kIN, wih0_bf, whh0_bf,
            h_buf, c_buf, pred_idx, pred_mask, step, 0,
            bih0, bhh0, h0o, c0o, x1_bf);

        // Layer 1: gates = h0@Wih1^T + mean(h_pred,l=1)@Whh1^T
        lstm_step_fused<<<lgrid, T, 0, stream>>>(
            x1_bf, kH, kH, wih1_bf, whh1_bf,
            h_buf, c_buf, pred_idx, pred_mask, step, 1,
            bih1, bhh1, h1o, c1o, nullptr);
    }

    // MLP head: fc = [h_buf[N, layer1] | features]
    const __bf16* last = h_buf + (size_t)kN * slotsz + (size_t)kB * kH;
    const dim3 mgrid(kB / 128, kMH / 64);
    mlp_gemm_relu_wmma<<<mgrid, T, 0, stream>>>(
        last, kH, kH, mw0_bf, kH + kEXTRA,
        feat_bf, kEXTRA, kEXTRA, mw0_bf + kH, kH + kEXTRA,
        mb0, act0_bf, kMH);
    mlp_gemm_relu_wmma<<<mgrid, T, 0, stream>>>(
        act0_bf, kMH, kMH, mw1_bf, kMH,
        nullptr, 0, 0, nullptr, 0,
        mb1, act1_bf, kMH);
    mlp_out_kernel<<<(kB + T - 1) / T, T, 0, stream>>>(act1_bf, mW2, mb2, out);
}
